// MolDecoder_5987184410770
// MI455X (gfx1250) — matrix operand order repaired and checked by reading, not compiled, not run on hardware
//
#include <hip/hip_runtime.h>

// ---------------- types ----------------
typedef __attribute__((ext_vector_type(16))) __bf16  v16bf;
typedef __attribute__((ext_vector_type(8)))  float   v8f;
typedef __attribute__((ext_vector_type(4)))  unsigned int v4u;

union Frag {
    v16bf v;
    v4u   q[2];
    unsigned short s[16];
};

// ---------------- packed-weight layout in d_ws (bf16 elements) ----------------
// B-tile (32x16) layout for v_wmma_f32_16x16x32_bf16:
//   lane<16 : element j -> B[k = kt*32 + (j<8 ? j : j+8), n = nt*16 + lane]
//   lane>=16: element j -> B[k = kt*32 + (j<8 ? j+8 : j+16), n = nt*16 + lane-16]
// tile (nt,kt) stored contiguously: 512 bf16, per-lane chunk = lane*16.
#define SZ_Z2H   (48 * 4 * 512)   // z2h_w [768,128]
#define SZ_WI0   (48 * 2 * 512)   // w_ih_0 [768,64]
#define SZ_WHH   (48 * 8 * 512)   // [768,256] matrices
#define SZ_OUTW  (4  * 8 * 512)   // out_w [64,256]
#define OFF_Z2H  0
#define OFF_WI0  (OFF_Z2H + SZ_Z2H)
#define OFF_WH0  (OFF_WI0 + SZ_WI0)
#define OFF_WI1  (OFF_WH0 + SZ_WHH)
#define OFF_WH1  (OFF_WI1 + SZ_WHH)
#define OFF_WI2  (OFF_WH1 + SZ_WHH)
#define OFF_WH2  (OFF_WI2 + SZ_WHH)
#define OFF_OUTW (OFF_WH2 + SZ_WHH)
#define OFF_EMB  (OFF_OUTW + SZ_OUTW)   // plain row-major [64][64] bf16

// ---------------- helpers ----------------
static __device__ __forceinline__ unsigned short f2bf(float f) {
    union { float f; unsigned int u; } v; v.f = f;
    unsigned int u = v.u;
    unsigned int r = (u + 0x7FFFu + ((u >> 16) & 1u)) >> 16;
    return (unsigned short)r;
}
static __device__ __forceinline__ float bf2f(unsigned short b) {
    union { unsigned int u; float f; } v; v.u = ((unsigned int)b) << 16;
    return v.f;
}
// branch-free activations: v_exp_f32 + v_rcp_f32, no saveexec blobs
static __device__ __forceinline__ float sigmoid_fast(float x) {
    return __builtin_amdgcn_rcpf(1.0f + __expf(-x));
}
static __device__ __forceinline__ float tanh_fast(float x) {
    float xc = fminf(fmaxf(x, -12.0f), 12.0f);
    float e  = __expf(2.0f * xc);
    return (e - 1.0f) * __builtin_amdgcn_rcpf(e + 1.0f);
}
static __device__ __forceinline__ v8f bcast8(float x) {
    v8f v;
    #pragma unroll
    for (int i = 0; i < 8; ++i) v[i] = x;
    return v;
}
static __device__ __forceinline__ v16bf loadB(const unsigned short* tile, int lane) {
    return *(const v16bf*)(tile + lane * 16);   // 32B contiguous per lane
}
static __device__ __forceinline__ v8f wmma_bf16(v16bf a, v16bf b, v8f c) {
    return __builtin_amdgcn_wmma_f32_16x16x32_bf16(false, a, false, b, (short)0, c,
                                                   false, false);
}
// A-fragment (16x32 bf16) from a [16][256] bf16 row-major buffer:
//   lane<16 : row m=lane,    K = kt*32 + {0..7, 16..23}
//   lane>=16: row m=lane-16, K = kt*32 + {8..15, 24..31}
static __device__ __forceinline__ void load_afrags8(const unsigned short* hl,
                                                    int m, int kbase, Frag* a) {
    const unsigned short* hr = hl + m * 256;
    #pragma unroll
    for (int kt = 0; kt < 8; ++kt) {
        a[kt].q[0] = *(const v4u*)(hr + kt * 32 + kbase);
        a[kt].q[1] = *(const v4u*)(hr + kt * 32 + kbase + 16);
    }
}

// 3 independent accumulator chains (one per gate) with distance-1 software
// pipelining of the B-fragment loads so loads overlap the WMMA chain.
template <int NK>
static __device__ __forceinline__ void mm3(const Frag* a,
                                           const unsigned short* __restrict__ wt,
                                           int nt, int lane,
                                           v8f& c0, v8f& c1, v8f& c2) {
    Frag b0, b1, b2, n0, n1, n2;
    b0.v = loadB(wt + (size_t)((0 * 16 + nt) * NK) * 512, lane);
    b1.v = loadB(wt + (size_t)((1 * 16 + nt) * NK) * 512, lane);
    b2.v = loadB(wt + (size_t)((2 * 16 + nt) * NK) * 512, lane);
    #pragma unroll
    for (int kt = 0; kt < NK; ++kt) {
        if (kt + 1 < NK) {
            n0.v = loadB(wt + (size_t)((0 * 16 + nt) * NK + kt + 1) * 512, lane);
            n1.v = loadB(wt + (size_t)((1 * 16 + nt) * NK + kt + 1) * 512, lane);
            n2.v = loadB(wt + (size_t)((2 * 16 + nt) * NK + kt + 1) * 512, lane);
        }
        c0 = wmma_bf16(a[kt].v, b0.v, c0);
        c1 = wmma_bf16(a[kt].v, b1.v, c1);
        c2 = wmma_bf16(a[kt].v, b2.v, c2);
        if (kt + 1 < NK) { b0 = n0; b1 = n1; b2 = n2; }
    }
}

// ---------------- GRU layer: 16-row tile, 4 waves (4 n-tiles each) ----------------
template <int NKI>
static __device__ __forceinline__ void gru_layer(
    int lane, int wave, const Frag* ax,
    const unsigned short* __restrict__ wiT,   // packed, 48 n-tiles x NKI k-tiles
    const unsigned short* __restrict__ whT,   // packed, 48 n-tiles x 8 k-tiles
    const float* biasI, const float* biasH,   // LDS, 768 each
    unsigned short* hl)                       // LDS [16][256] bf16 (old h -> new h)
{
    const int m      = lane & 15;
    const int kbase  = (lane < 16) ? 0 : 8;
    const int colL   = lane & 15;
    const int mmBase = (lane < 16) ? 0 : 8;

    Frag ah[8];
    load_afrags8(hl, m, kbase, ah);   // capture old h
    __syncthreads();                  // all waves captured old h before overwrite

    #pragma unroll 1
    for (int nt0 = 0; nt0 < 4; ++nt0) {
        const int nt  = wave * 4 + nt0;
        const int col = nt * 16 + colL;
        if (nt0 < 3) {  // prefetch next n-tile's hh weights (global_prefetch)
            __builtin_prefetch(whT + (size_t)((0 * 16 + nt + 1) * 8) * 512, 0, 0);
            __builtin_prefetch(whT + (size_t)((1 * 16 + nt + 1) * 8) * 512, 0, 0);
            __builtin_prefetch(whT + (size_t)((2 * 16 + nt + 1) * 8) * 512, 0, 0);
        }
        v8f air = bcast8(biasI[col]);
        v8f aiz = bcast8(biasI[256 + col]);
        v8f ain = bcast8(biasI[512 + col]);
        v8f ahr = bcast8(biasH[col]);
        v8f ahz = bcast8(biasH[256 + col]);
        v8f ahn = bcast8(biasH[512 + col]);
        mm3<NKI>(ax, wiT, nt, lane, air, aiz, ain);
        mm3<8>(ah, whT, nt, lane, ahr, ahz, ahn);
        #pragma unroll
        for (int r = 0; r < 8; ++r) {
            float rg = sigmoid_fast(air[r] + ahr[r]);
            float ug = sigmoid_fast(aiz[r] + ahz[r]);
            float ng = tanh_fast(ain[r] + rg * ahn[r]);
            int   mm = mmBase + r;
            float ho = bf2f(hl[mm * 256 + col]);
            float hn = (1.0f - ug) * ng + ug * ho;
            hl[mm * 256 + col] = f2bf(hn);
        }
    }
    __syncthreads();                  // new h complete (next layer reads it as x)
}

// ---------------- pack kernels ----------------
__global__ void pack_tiled(const float* __restrict__ W, unsigned short* __restrict__ dst,
                           int K, int ktiles, int total) {
    int e = blockIdx.x * blockDim.x + threadIdx.x;
    if (e >= total) return;
    int tile = e >> 9, r = e & 511;
    int lane = r >> 4, j = r & 15;
    int nt = tile / ktiles, kt = tile % ktiles;
    int n = nt * 16 + (lane & 15);
    int k = kt * 32 + ((lane >= 16) ? 8 : 0) + (j & 7) + ((j >= 8) ? 16 : 0);
    dst[e] = f2bf(W[(size_t)n * K + k]);
}
__global__ void cvt_bf16(const float* __restrict__ src, unsigned short* __restrict__ dst, int n) {
    int i = blockIdx.x * blockDim.x + threadIdx.x;
    if (i < n) dst[i] = f2bf(src[i]);
}

// ---------------- persistent decoder: 4 waves per 16 batch rows ----------------
__global__ __launch_bounds__(128) void moldec_kernel(
    const float* __restrict__ z,
    const float* __restrict__ z2h_b,
    const float* __restrict__ bih0, const float* __restrict__ bhh0,
    const float* __restrict__ bih1, const float* __restrict__ bhh1,
    const float* __restrict__ bih2, const float* __restrict__ bhh2,
    const float* __restrict__ outb,
    const unsigned short* __restrict__ wsp,
    float* __restrict__ outp)
{
    __shared__ __align__(16) unsigned short hbuf[3 * 16 * 256]; // h per layer, bf16
    __shared__ __align__(16) float lbuf[16 * 64];               // logits for argmax
    __shared__ float biasI[3][768];
    __shared__ float biasH[3][768];
    __shared__ float biasO[64];
    __shared__ int   tokbuf[16];

    const int tid      = threadIdx.x;
    const int lane     = tid & 31;
    const int wave     = tid >> 5;        // 0..3
    const int tileBase = blockIdx.x * 16;
    const int m        = lane & 15;
    const int kbase    = (lane < 16) ? 0 : 8;
    const int colL     = lane & 15;
    const int mmBase   = (lane < 16) ? 0 : 8;

    for (int i = tid; i < 768; i += 128) {
        biasI[0][i] = bih0[i]; biasH[0][i] = bhh0[i];
        biasI[1][i] = bih1[i]; biasH[1][i] = bhh1[i];
        biasI[2][i] = bih2[i]; biasH[2][i] = bhh2[i];
    }
    if (tid < 64) biasO[tid] = outb[tid];
    if (tid < 16) tokbuf[tid] = 1;        // start token
    __syncthreads();

    // ---- h0 = tanh(z @ z2h_w.T + z2h_b), reshaped to 3 x [16,256] ----
    Frag az[4];
    {
        const float* zr = z + (size_t)(tileBase + m) * 128;
        #pragma unroll
        for (int kt = 0; kt < 4; ++kt) {
            #pragma unroll
            for (int j = 0; j < 8; ++j) az[kt].s[j]     = f2bf(zr[kt * 32 + kbase + j]);
            #pragma unroll
            for (int j = 0; j < 8; ++j) az[kt].s[8 + j] = f2bf(zr[kt * 32 + kbase + 16 + j]);
        }
    }
    #pragma unroll 1
    for (int ng0 = 0; ng0 < 12; ++ng0) {  // 48 n-tiles split over 4 waves
        const int ng = wave * 12 + ng0;
        v8f acc = bcast8(z2h_b[ng * 16 + colL]);
        #pragma unroll
        for (int kt = 0; kt < 4; ++kt)
            acc = wmma_bf16(az[kt].v, loadB(wsp + OFF_Z2H + (size_t)(ng * 4 + kt) * 512, lane), acc);
        unsigned short* hl = hbuf + (ng >> 4) * 4096;
        int hcol = (ng & 15) * 16 + colL;
        #pragma unroll
        for (int r = 0; r < 8; ++r)
            hl[(mmBase + r) * 256 + hcol] = f2bf(tanh_fast(acc[r]));
    }
    __syncthreads();

    const unsigned short* embbf = wsp + OFF_EMB;

    #pragma unroll 1
    for (int t = 0; t < 119; ++t) {
        // layer 0 input: x = emb[tok]  (bf16 rows, gathered per lane)
        Frag ax0[2];
        {
            int tr = tokbuf[m];
            const unsigned short* er = embbf + (size_t)tr * 64;
            #pragma unroll
            for (int kt = 0; kt < 2; ++kt) {
                ax0[kt].q[0] = *(const v4u*)(er + kt * 32 + kbase);
                ax0[kt].q[1] = *(const v4u*)(er + kt * 32 + kbase + 16);
            }
        }
        gru_layer<2>(lane, wave, ax0, wsp + OFF_WI0, wsp + OFF_WH0, biasI[0], biasH[0], hbuf);

        Frag ax[8];
        load_afrags8(hbuf, m, kbase, ax);                       // new h0 -> x of layer 1
        gru_layer<8>(lane, wave, ax, wsp + OFF_WI1, wsp + OFF_WH1, biasI[1], biasH[1], hbuf + 4096);

        load_afrags8(hbuf + 4096, m, kbase, ax);                // new h1 -> x of layer 2
        gru_layer<8>(lane, wave, ax, wsp + OFF_WI2, wsp + OFF_WH2, biasI[2], biasH[2], hbuf + 8192);

        // output projection: logits = h2 @ out_w.T + out_b   (ot = wave)
        load_afrags8(hbuf + 8192, m, kbase, ax);
        {
            const int ot  = wave;
            const int col = ot * 16 + colL;
            v8f acc = bcast8(biasO[col]);
            Frag bc, bn;
            bc.v = loadB(wsp + OFF_OUTW + (size_t)(ot * 8) * 512, lane);
            #pragma unroll
            for (int kt = 0; kt < 8; ++kt) {
                if (kt + 1 < 8)
                    bn.v = loadB(wsp + OFF_OUTW + (size_t)(ot * 8 + kt + 1) * 512, lane);
                acc = wmma_bf16(ax[kt].v, bc.v, acc);
                if (kt + 1 < 8) bc = bn;
            }
            #pragma unroll
            for (int r = 0; r < 8; ++r) {
                int mm = mmBase + r;
                float v = acc[r];
                lbuf[mm * 64 + col] = v;
                outp[((size_t)(tileBase + mm) * 119 + t) * 64 + col] = v;
            }
        }
        __syncthreads();
        // argmax per row (first max, matches jnp.argmax)
        if (wave == 0 && lane < 16) {
            float best = lbuf[lane * 64];
            int bidx = 0;
            #pragma unroll 1
            for (int v = 1; v < 64; ++v) {
                float f = lbuf[lane * 64 + v];
                if (f > best) { best = f; bidx = v; }
            }
            tokbuf[lane] = bidx;
        }
        __syncthreads();
    }
}

// ---------------- launch ----------------
extern "C" void kernel_launch(void* const* d_in, const int* in_sizes, int n_in,
                              void* d_out, int out_size, void* d_ws, size_t ws_size,
                              hipStream_t stream) {
    (void)in_sizes; (void)n_in; (void)out_size; (void)ws_size;
    const float* z     = (const float*)d_in[0];
    const float* emb   = (const float*)d_in[1];
    const float* z2h_w = (const float*)d_in[2];
    const float* z2h_b = (const float*)d_in[3];
    const float* out_w = (const float*)d_in[4];
    const float* out_b = (const float*)d_in[5];
    const float* wih0  = (const float*)d_in[6];
    const float* whh0  = (const float*)d_in[7];
    const float* bih0  = (const float*)d_in[8];
    const float* bhh0  = (const float*)d_in[9];
    const float* wih1  = (const float*)d_in[10];
    const float* whh1  = (const float*)d_in[11];
    const float* bih1  = (const float*)d_in[12];
    const float* bhh1  = (const float*)d_in[13];
    const float* wih2  = (const float*)d_in[14];
    const float* whh2  = (const float*)d_in[15];
    const float* bih2  = (const float*)d_in[16];
    const float* bhh2  = (const float*)d_in[17];
    unsigned short* wsp = (unsigned short*)d_ws;
    float* outp = (float*)d_out;

    auto pack = [&](const float* W, size_t off, int K, int ktiles, int total) {
        pack_tiled<<<(total + 255) / 256, 256, 0, stream>>>(W, wsp + off, K, ktiles, total);
    };
    pack(z2h_w, OFF_Z2H, 128, 4, SZ_Z2H);
    pack(wih0,  OFF_WI0,  64, 2, SZ_WI0);
    pack(whh0,  OFF_WH0, 256, 8, SZ_WHH);
    pack(wih1,  OFF_WI1, 256, 8, SZ_WHH);
    pack(whh1,  OFF_WH1, 256, 8, SZ_WHH);
    pack(wih2,  OFF_WI2, 256, 8, SZ_WHH);
    pack(whh2,  OFF_WH2, 256, 8, SZ_WHH);
    pack(out_w, OFF_OUTW, 256, 8, SZ_OUTW);
    cvt_bf16<<<(4096 + 255) / 256, 256, 0, stream>>>(emb, wsp + OFF_EMB, 4096);

    moldec_kernel<<<256, 128, 0, stream>>>(z, z2h_b,
                                           bih0, bhh0, bih1, bhh1, bih2, bhh2,
                                           out_b, wsp, outp);
}
